// MHA_86835648791002
// MI455X (gfx1250) — compile-verified, matrix operand order checked by reading, not run on hardware
//
#include <hip/hip_runtime.h>
#include <math.h>

// ---------------- problem constants ----------------
constexpr int Bc   = 2;
constexpr int Sc   = 2048;
constexpr int Dm   = 2048;
constexpr int HQ   = 16;
constexpr int HKV  = 4;
constexpr int HD   = 128;
constexpr int REP  = HQ / HKV;          // 4
constexpr float SCALE = 0.022097086912079612f;  // 1/sqrt(2048)

typedef __attribute__((ext_vector_type(16))) __bf16   v16bf;
typedef __attribute__((ext_vector_type(8)))  float    v8f;
typedef __attribute__((ext_vector_type(4)))  float    fvec4;
typedef __attribute__((ext_vector_type(2)))  float    fvec2;
typedef __attribute__((ext_vector_type(4)))  unsigned uvec4;
typedef __attribute__((ext_vector_type(8)))  int      ivec8;
typedef __attribute__((ext_vector_type(4)))  int      ivec4;

union Frag { v16bf v; unsigned u[8]; };

__device__ __forceinline__ unsigned pack_bf16(float a, float b) {
    unsigned short ua = __builtin_bit_cast(unsigned short, (__bf16)a);
    unsigned short ub = __builtin_bit_cast(unsigned short, (__bf16)b);
    return (unsigned)ua | ((unsigned)ub << 16);
}

// u32 (K-pair) index within a 16-bit 16x32 WMMA fragment, vgpr j (0..7):
// K = ((j<4)?0:16) + (lane_hi?8:0) + 2*(j&3) + p   ->  u32 index = K/2
__device__ __forceinline__ int pair_idx(int j, int lane_hi) {
    return ((j < 4) ? 0 : 8) + (lane_hi ? 4 : 0) + (j & 3);
}

// 16x32 fragment from a K-contiguous bf16 row: two 16B loads, no conversion.
__device__ __forceinline__ void frag_from_bf16(Frag& f, const __bf16* row, int kk, int lane_hi) {
    uvec4 t0 = *(const uvec4*)(row + kk + lane_hi * 8);
    uvec4 t1 = *(const uvec4*)(row + kk + 16 + lane_hi * 8);
    f.u[0] = t0.x; f.u[1] = t0.y; f.u[2] = t0.z; f.u[3] = t0.w;
    f.u[4] = t1.x; f.u[5] = t1.y; f.u[6] = t1.z; f.u[7] = t1.w;
}

// 16x32 fragment from a K-contiguous fp32 row (nontemporal fvec4 loads + pack).
__device__ __forceinline__ void frag_from_f32_nt(Frag& f, const fvec4* r4, int kbase, int lane_hi) {
    #pragma unroll
    for (int h4 = 0; h4 < 2; ++h4) {
        int fi = (kbase >> 2) + h4 * 4 + lane_hi * 2;
        fvec4 a0 = __builtin_nontemporal_load(r4 + fi);
        fvec4 a1 = __builtin_nontemporal_load(r4 + fi + 1);
        f.u[h4 * 4 + 0] = pack_bf16(a0.x, a0.y);
        f.u[h4 * 4 + 1] = pack_bf16(a0.z, a0.w);
        f.u[h4 * 4 + 2] = pack_bf16(a1.x, a1.y);
        f.u[h4 * 4 + 3] = pack_bf16(a1.z, a1.w);
    }
}

// ---------------- GEMM: C[M,N] = A[M,K] @ W[K,N] (+bias), bf16 WMMA, LDS ping-pong ----------------
#define TM 64
#define TN 128
#define TK 32
#define LDP 17   // padded u32 row stride: conflict-free 16-row column reads

template <bool HAS_BIAS>
__global__ __launch_bounds__(256) void gemm_bf16(const float* __restrict__ A,
                                                 const float* __restrict__ W,
                                                 const float* __restrict__ bias,
                                                 float* __restrict__ C,
                                                 int M, int N, int K) {
    __shared__ unsigned As32[2][TM * LDP];   // [row][k/2] packed bf16 pairs along K
    __shared__ unsigned Bs32[2][TN * LDP];   // [col][k/2] packed bf16 pairs along K

    const int tid     = threadIdx.x;
    const int lane    = tid & 31;
    const int wave    = tid >> 5;
    const int wm      = wave & 3;
    const int wn      = wave >> 2;
    const int lane_hi = lane >> 4;
    const int lane_lo = lane & 15;
    const int block_m = blockIdx.y * TM;
    const int block_n = blockIdx.x * TN;

    fvec2 ra[4];                 // raw A staging (pack deferred to LDS store)
    float rb0[8], rb1[8];        // raw B staging

    auto load_tile = [&](int k0) {
        #pragma unroll
        for (int e = 0; e < 4; ++e) {
            int idx = e * 256 + tid;              // 0..1023
            int r = idx >> 4, c2 = idx & 15;
            ra[e] = *(const fvec2*)(A + (size_t)(block_m + r) * K + k0 + c2 * 2);
        }
        #pragma unroll
        for (int e = 0; e < 8; ++e) {
            int idx = e * 256 + tid;              // 0..2047
            int r2 = idx >> 7, c = idx & 127;
            const float* wp = W + (size_t)(k0 + r2 * 2) * N + block_n + c;
            rb0[e] = wp[0];
            rb1[e] = wp[N];
        }
    };
    auto store_tile = [&](int buf) {
        #pragma unroll
        for (int e = 0; e < 4; ++e) {
            int idx = e * 256 + tid;
            As32[buf][(idx >> 4) * LDP + (idx & 15)] = pack_bf16(ra[e].x, ra[e].y);
        }
        #pragma unroll
        for (int e = 0; e < 8; ++e) {
            int idx = e * 256 + tid;
            Bs32[buf][(idx & 127) * LDP + (idx >> 7)] = pack_bf16(rb0[e], rb1[e]);
        }
    };

    v8f acc[4] = {};

    load_tile(0);
    store_tile(0);
    __syncthreads();

    int buf = 0;
    for (int k0 = 0; k0 < K; k0 += TK) {
        // unconditional clamped prefetch: deterministic load counts -> partial waits
        const int k_next = (k0 + TK < K) ? (k0 + TK) : k0;
        load_tile(k_next);

        Frag fa;
        #pragma unroll
        for (int j = 0; j < 8; ++j)
            fa.u[j] = As32[buf][(wm * 16 + lane_lo) * LDP + pair_idx(j, lane_hi)];
        #pragma unroll
        for (int s = 0; s < 4; ++s) {
            Frag fb;
            #pragma unroll
            for (int j = 0; j < 8; ++j)
                fb.u[j] = Bs32[buf][(wn * 64 + s * 16 + lane_lo) * LDP + pair_idx(j, lane_hi)];
            acc[s] = __builtin_amdgcn_wmma_f32_16x16x32_bf16(false, fa.v, false, fb.v,
                                                             (short)0, acc[s], false, false);
        }

        store_tile(buf ^ 1);                  // waits for loads here, after compute
        __syncthreads();
        buf ^= 1;
    }

    #pragma unroll
    for (int s = 0; s < 4; ++s) {
        #pragma unroll
        for (int j = 0; j < 8; ++j) {
            int row = block_m + wm * 16 + j + lane_hi * 8;
            int col = block_n + wn * 64 + s * 16 + lane_lo;
            float bb = HAS_BIAS ? bias[col] : 0.0f;
            C[(size_t)row * N + col] = acc[s][j] + bb;
        }
    }
}

// ---------------- RoPE fp32 -> packed bf16 (pair == one u32), optional output scale ----------------
__global__ void rope_to_bf16(const float* __restrict__ xin, __bf16* __restrict__ xout,
                             const float* __restrict__ cs, const float* __restrict__ sn,
                             int H, int total, float oscale) {
    int idx = blockIdx.x * blockDim.x + threadIdx.x;
    if (idx >= total) return;
    const int half = HD / 2;
    int d2 = idx % half;
    int h  = (idx / half) % H;
    int s  = (idx / (half * H)) % Sc;
    int b  =  idx / (half * H * Sc);
    size_t base = (((size_t)b * Sc + s) * H + h) * HD;
    float xr = xin[base + 2 * d2];
    float xi = xin[base + 2 * d2 + 1];
    float c  = cs[s * half + d2];
    float si = sn[s * half + d2];
    *(unsigned*)(xout + base + 2 * d2) =
        pack_bf16((xr * c - xi * si) * oscale, (xr * si + xi * c) * oscale);
}

// ---------------- v fp32 (B,S,HKV,HD) -> vT bf16 (B,HKV,HD,S) ----------------
// Staging via Tensor Data Mover: 2D tile 128x64 fp32, row stride 512, LDS row pad
// 1 dword per 128 dwords -> 129-float LDS stride (conflict-free transpose reads).
#define VTP 129

__global__ __launch_bounds__(256) void v_to_bf16T(const float* __restrict__ v,
                                                  __bf16* __restrict__ vT) {
    __shared__ float tile[64 * VTP];      // LDS offset 0 (only LDS object here)
    const int kc  = blockIdx.x;           // S/64 chunks
    const int bkv = blockIdx.y;           // b*HKV + kvh
    const int b   = bkv / HKV;
    const int kvh = bkv % HKV;
    const int tid = threadIdx.x;

    const size_t gbase = (size_t)(((size_t)b * Sc + kc * 64) * HKV + kvh) * HD;

#if __has_builtin(__builtin_amdgcn_tensor_load_to_lds) && __has_builtin(__builtin_amdgcn_s_wait_tensorcnt)
    if (tid < 32) {                       // one wave issues the TDM descriptor
        unsigned long long ga = (unsigned long long)(const void*)(v + gbase);
        uvec4 g0;
        g0.x = 1u;                                        // count=1, user descriptor
        g0.y = 0u;                                        // lds_addr = 0
        g0.z = (unsigned)(ga & 0xFFFFFFFFull);            // global_addr[31:0]
        g0.w = (unsigned)((ga >> 32) & 0x01FFFFFFull)     // global_addr[56:32]
             | 0x80000000u;                               // type = 2 ("image")
        ivec8 g1;
        g1[0] = (int)((2u << 16)                          // data_size = 4B
             | (1u << 20)                                 // pad_enable
             | (6u << 22));                               // pad_interval: 128 dwords (+1 dword)
        g1[1] = (int)(((unsigned)HD & 0xFFFFu) << 16);    // tensor_dim0 = 128 (low 16)
        g1[2] = (int)(64u << 16);                         // tensor_dim0 hi=0 | tensor_dim1 = 64
        g1[3] = (int)(((unsigned)HD) << 16);              // tensor_dim1 hi=0 | tile_dim0 = 128
        g1[4] = 64;                                       // tile_dim1 = 64, tile_dim2 = 0
        g1[5] = HKV * HD;                                 // tensor_dim0_stride = 512
        g1[6] = 0;
        g1[7] = 0;
        ivec4 g2 = {0, 0, 0, 0};
        ivec4 g3 = {0, 0, 0, 0};
        ivec8 g4 = {0, 0, 0, 0, 0, 0, 0, 0};
        __builtin_amdgcn_tensor_load_to_lds(g0, g1, g2, g3, g4, 0);
        __builtin_amdgcn_s_wait_tensorcnt(0);
    }
    __syncthreads();
#else
    #pragma unroll
    for (int e = 0; e < 8; ++e) {
        int idx = e * 256 + tid;          // 0..2047 fvec4s
        int r = idx >> 5, c4 = (idx & 31) * 4;
        fvec4 t = *(const fvec4*)(v + gbase + (size_t)r * (HKV * HD) + c4);
        tile[r * VTP + c4 + 0] = t.x;
        tile[r * VTP + c4 + 1] = t.y;
        tile[r * VTP + c4 + 2] = t.z;
        tile[r * VTP + c4 + 3] = t.w;
    }
    __syncthreads();
#endif

    #pragma unroll
    for (int e = 0; e < 16; ++e) {
        int idx = e * 256 + tid;          // 0..4095 u32s
        int c   = idx >> 5;               // col 0..127
        int k2  = idx & 31;               // u32 along k
        float lo = tile[(k2 * 2    ) * VTP + c];
        float hi = tile[(k2 * 2 + 1) * VTP + c];
        *(unsigned*)(vT + ((size_t)bkv * HD + c) * Sc + kc * 64 + k2 * 2) = pack_bf16(lo, hi);
    }
}

// ---------------- scores helpers ----------------
__device__ __forceinline__ void load_kfrags(Frag kf[4], const __bf16* krow, int lane_hi) {
    #pragma unroll
    for (int s4 = 0; s4 < 4; ++s4) frag_from_bf16(kf[s4], krow, s4 * 32, lane_hi);
}

__device__ __forceinline__ void score_tile(const Frag qf[4], const Frag kf[4],
                                           float* __restrict__ attn,
                                           int bh, int qt, int kt, int lane_hi, int lane_lo) {
    v8f acc = {};
    #pragma unroll
    for (int s4 = 0; s4 < 4; ++s4)
        acc = __builtin_amdgcn_wmma_f32_16x16x32_bf16(false, qf[s4].v, false, kf[s4].v,
                                                      (short)0, acc, false, false);
    // q was pre-scaled by 1/sqrt(D) at RoPE time: store accumulators directly
    #pragma unroll
    for (int j = 0; j < 8; ++j) {
        int row = qt * 16 + j + lane_hi * 8;
        int col = kt * 16 + lane_lo;
        __builtin_nontemporal_store(acc[j], attn + ((size_t)bh * Sc + row) * Sc + col);
    }
}

// ---------------- scores: one wave per (q tile, 16-kt strip), double-buffered k fragments ----------------
__global__ __launch_bounds__(32) void attn_scores(const __bf16* __restrict__ qbf,
                                                  const __bf16* __restrict__ kbf,
                                                  float* __restrict__ attn) {
    const int qt  = blockIdx.x;
    const int kc  = blockIdx.y;           // strip of 16 kt tiles
    const int bh  = blockIdx.z;
    const int kt0 = kc * 16;
    if (kt0 > qt) return;
    const int kt1 = (kt0 + 16 < qt + 1) ? kt0 + 16 : qt + 1;

    const int b   = bh / HQ;
    const int h   = bh % HQ;
    const int kvh = h / REP;
    const int lane    = threadIdx.x & 31;
    const int lane_hi = lane >> 4;
    const int lane_lo = lane & 15;

    const __bf16* qrow  = qbf + (((size_t)b * Sc + qt * 16 + lane_lo) * HQ + h) * HD;
    const __bf16* krow0 = kbf + (((size_t)b * Sc + lane_lo) * HKV + kvh) * HD;
    const size_t  kstep = (size_t)16 * HKV * HD;    // bf16 elements per kt tile

    Frag qf[4];
    #pragma unroll
    for (int s4 = 0; s4 < 4; ++s4) frag_from_bf16(qf[s4], qrow, s4 * 32, lane_hi);

    Frag kfA[4], kfB[4];
    load_kfrags(kfA, krow0 + (size_t)kt0 * kstep, lane_hi);

    for (int kt = kt0; kt < kt1; kt += 2) {
        // unconditional clamped prefetches keep load counts deterministic
        const int ktB = (kt + 1 < kt1) ? kt + 1 : kt;
        load_kfrags(kfB, krow0 + (size_t)ktB * kstep, lane_hi);

        score_tile(qf, kfA, attn, bh, qt, kt, lane_hi, lane_lo);

        const int ktA = (kt + 2 < kt1) ? kt + 2 : kt;
        load_kfrags(kfA, krow0 + (size_t)ktA * kstep, lane_hi);

        if (kt + 1 < kt1)
            score_tile(qf, kfB, attn, bh, qt, kt + 1, lane_hi, lane_lo);
    }
}

// ---------------- row softmax with causal zero-fill ----------------
__global__ __launch_bounds__(256) void softmax_rows(float* __restrict__ attn) {
    __shared__ float red[256];
    const int tid = threadIdx.x;
    const size_t row = blockIdx.x;          // bh*S + i
    const int i = (int)(row % Sc);
    float* p = attn + row * (size_t)Sc;
    const int n = i + 1;

    float m = -INFINITY;
    for (int j = tid; j < n; j += 256) m = fmaxf(m, p[j]);
    red[tid] = m; __syncthreads();
    for (int s2 = 128; s2 > 0; s2 >>= 1) {
        if (tid < s2) red[tid] = fmaxf(red[tid], red[tid + s2]);
        __syncthreads();
    }
    m = red[0]; __syncthreads();

    float sum = 0.0f;
    for (int j = tid; j < n; j += 256) sum += __expf(p[j] - m);
    red[tid] = sum; __syncthreads();
    for (int s2 = 128; s2 > 0; s2 >>= 1) {
        if (tid < s2) red[tid] += red[tid + s2];
        __syncthreads();
    }
    sum = red[0];
    float inv = 1.0f / sum;
    for (int j = tid; j < n; j += 256) {
        float val = __expf(p[j] - m) * inv;
        __builtin_nontemporal_store(val, p + j);
    }
    for (int j = n + tid; j < Sc; j += 256)
        __builtin_nontemporal_store(0.0f, p + j);
}

// ---------------- AV: one wave per (head, 16-row q tile); pipelined loads ----------------
__global__ __launch_bounds__(32) void attn_v(const float* __restrict__ attn,
                                             const __bf16* __restrict__ vT,
                                             float* __restrict__ ctx) {
    const int qt = blockIdx.x;
    const int bh = blockIdx.y;
    const int b   = bh / HQ;
    const int h   = bh % HQ;
    const int kvh = h / REP;
    const int lane    = threadIdx.x & 31;
    const int lane_hi = lane >> 4;
    const int lane_lo = lane & 15;

    v8f acc[8] = {};
    const fvec4* a4 = (const fvec4*)(attn + ((size_t)bh * Sc + qt * 16 + lane_lo) * Sc);
    const __bf16* vcol0 = vT + (((size_t)b * HKV + kvh) * HD + lane_lo) * Sc;
    const int nk = ((qt + 1) * 16 + 31) / 32;   // causal bound

    Frag fa, fan;
    frag_from_f32_nt(fa, a4, 0, lane_hi);

    for (int t = 0; t < nk; ++t) {
        const int kbase = t * 32;

        // all 16 v-fragment loads issued ahead of the WMMA chain (consumed in order)
        Frag fb[8];
        #pragma unroll
        for (int nt = 0; nt < 8; ++nt)
            frag_from_bf16(fb[nt], vcol0 + (size_t)(nt * 16) * Sc, kbase, lane_hi);

        // unconditional clamped prefetch of the next K-step's attn fragment
        const int kn = (t + 1 < nk) ? kbase + 32 : kbase;
        frag_from_f32_nt(fan, a4, kn, lane_hi);

        #pragma unroll
        for (int nt = 0; nt < 8; ++nt)
            acc[nt] = __builtin_amdgcn_wmma_f32_16x16x32_bf16(false, fa.v, false, fb[nt].v,
                                                              (short)0, acc[nt], false, false);
        fa = fan;
    }

    #pragma unroll
    for (int nt = 0; nt < 8; ++nt) {
        #pragma unroll
        for (int j = 0; j < 8; ++j) {
            int row = qt * 16 + j + lane_hi * 8;
            ctx[((size_t)b * Sc + row) * Dm + h * HD + nt * 16 + lane_lo] = acc[nt][j];
        }
    }
}

// ---------------- host launcher ----------------
extern "C" void kernel_launch(void* const* d_in, const int* in_sizes, int n_in,
                              void* d_out, int out_size, void* d_ws, size_t ws_size,
                              hipStream_t stream) {
    (void)in_sizes; (void)n_in; (void)out_size; (void)ws_size;
    const float* x    = (const float*)d_in[0];
    const float* fcos = (const float*)d_in[1];
    const float* fsin = (const float*)d_in[2];
    const float* wq   = (const float*)d_in[3];
    const float* bq   = (const float*)d_in[4];
    const float* wk   = (const float*)d_in[5];
    const float* bk   = (const float*)d_in[6];
    const float* wv   = (const float*)d_in[7];
    const float* bv   = (const float*)d_in[8];
    const float* wo   = (const float*)d_in[9];

    float* out  = (float*)d_out;
    float* attn = out + (size_t)Bc * Sc * Dm;            // (B,HQ,S,S)

    const int M = Bc * Sc;                               // 4096
    const size_t QE = (size_t)M * HQ * HD;               // 8.4M
    const size_t KE = (size_t)M * HKV * HD;              // 2.1M

    char* w = (char*)d_ws;
    float*  qf32 = (float*)w;              w += QE * sizeof(float);
    float*  kf32 = (float*)w;              w += KE * sizeof(float);
    float*  vf32 = (float*)w;              w += KE * sizeof(float);
    float*  ctx  = (float*)w;              w += QE * sizeof(float);
    __bf16* qbf  = (__bf16*)w;             w += QE * sizeof(__bf16);
    __bf16* kbf  = (__bf16*)w;             w += KE * sizeof(__bf16);
    __bf16* vT   = (__bf16*)w;

    // 1) QKV projections (fp32 out)
    gemm_bf16<true><<<dim3((HQ * HD) / TN, M / TM), 256, 0, stream>>>(x, wq, bq, qf32, M, HQ * HD, Dm);
    gemm_bf16<true><<<dim3((HKV * HD) / TN, M / TM), 256, 0, stream>>>(x, wk, bk, kf32, M, HKV * HD, Dm);
    gemm_bf16<true><<<dim3((HKV * HD) / TN, M / TM), 256, 0, stream>>>(x, wv, bv, vf32, M, HKV * HD, Dm);

    // 2) RoPE -> packed bf16 (q pre-scaled by 1/sqrt(D)); v -> bf16 K-major transpose (TDM-staged)
    {
        int tq = Bc * Sc * HQ * (HD / 2);
        rope_to_bf16<<<(tq + 255) / 256, 256, 0, stream>>>(qf32, qbf, fcos, fsin, HQ, tq, SCALE);
        int tk = Bc * Sc * HKV * (HD / 2);
        rope_to_bf16<<<(tk + 255) / 256, 256, 0, stream>>>(kf32, kbf, fcos, fsin, HKV, tk, 1.0f);
        v_to_bf16T<<<dim3(Sc / 64, Bc * HKV), 256, 0, stream>>>(vf32, vT);
    }

    // 3) causal scores (lower-triangular tiles only)
    attn_scores<<<dim3(Sc / 16, 8, Bc * HQ), 32, 0, stream>>>(qbf, kbf, attn);

    // 4) row softmax (writes zeros above the diagonal)
    softmax_rows<<<Bc * HQ * Sc, 256, 0, stream>>>(attn);

    // 5) ctx = attn @ v (causal-bounded K loop)
    attn_v<<<dim3(Sc / 16, Bc * HQ), 32, 0, stream>>>(attn, vT, ctx);

    // 6) out = ctx @ wo
    gemm_bf16<false><<<dim3(Dm / TN, M / TM), 256, 0, stream>>>(ctx, wo, nullptr, out, M, Dm, Dm);
}
